// QLSTMEnhanced_65481071409482
// MI455X (gfx1250) — matrix operand order repaired and checked by reading, not compiled, not run on hardware
//
#include <hip/hip_runtime.h>
#include <hip/hip_bf16.h>

typedef __attribute__((ext_vector_type(16))) __bf16 v16bf;
typedef __attribute__((ext_vector_type(8)))  float  v8f;

#define T_SEQ 512
#define BATCH 4096
#define DIM   64
#define HID   16
#define FFD   64

#define WAVES_PER_BLOCK 2
#define ROWS_PER_WAVE   16
#define ROWS_PER_BLOCK  (WAVES_PER_BLOCK * ROWS_PER_WAVE)

// D = SRC0(A) x SRC1(B) + C ; A = weights (fragment-ordered), B = activations (registers)
#define WMMA_BF16(A, B, C) \
  __builtin_amdgcn_wmma_f32_16x16x32_bf16(false, (A), false, (B), (short)0, (C), false, false)

union FragU { v16bf v; unsigned int u[8]; };

__device__ __forceinline__ unsigned int pkbf(float a, float b) {
  union { __bf16 h[2]; unsigned int u; } x;
  x.h[0] = (__bf16)a; x.h[1] = (__bf16)b;
  return x.u;
}

// gfx1250 hardware transcendentals (v_tanh_f32 / v_exp_f32 / v_rcp_f32 / v_cos_f32)
#if __has_builtin(__builtin_amdgcn_tanhf)
__device__ __forceinline__ float fast_tanh(float x) { return __builtin_amdgcn_tanhf(x); }
#else
__device__ __forceinline__ float fast_tanh(float x) {
  float e2 = __expf(2.0f * x);                       // v_exp_f32 based
  return 1.0f - 2.0f * __builtin_amdgcn_rcpf(e2 + 1.0f);
}
#endif
__device__ __forceinline__ float fast_sigmoid(float x) {
  return __builtin_amdgcn_rcpf(1.0f + __expf(-x));
}

// bf16(1.0) in low half of a dword (multiplies the bias row of the weight A-fragment)
#define BF16_ONE_LO 0x00003F80u

__global__ __launch_bounds__(WAVES_PER_BLOCK * 32)
void qlstm_scan_kernel(const float* __restrict__ inputs,
                       const float* __restrict__ Wg,  const float* __restrict__ bg,
                       const float* __restrict__ W1,  const float* __restrict__ b1,
                       const float* __restrict__ W2,  const float* __restrict__ b2,
                       const float* __restrict__ theta,
                       float* __restrict__ out)
{
  // Weight A-fragments, fragment-ordered (read-only after init; no fences needed).
  // A-frag kl pattern: kl = (lane>=16?16:0) + j
  __shared__ __align__(32) __bf16 sWgA[4 * 3 * 32 * 16]; // [g][kstep 96/32][lane][16], k==80 row = bg
  __shared__ __align__(32) __bf16 sW1A[4 * 4 * 32 * 16]; // [g][ftile 64/16][lane][16], k==16 row = b1
  __shared__ __align__(32) __bf16 sW2A[4 * 2 * 32 * 16]; // [g][kstep 64/32][lane][16]

  const int tid  = threadIdx.x;
  const int wave = tid >> 5;
  const int lane = tid & 31;
  const bool hi  = lane >= 16;
  const int b    = lane & 15;                 // batch row owned by this lane (both halves)
  const int hiOff = hi ? 8 : 0;               // feature offset held in VGPRs (D layout)
  const int rowBase = blockIdx.x * ROWS_PER_BLOCK + wave * ROWS_PER_WAVE;

  // ---------------- init: swizzle weights (+fold biases) into A-fragment order --------
  for (int idx = tid; idx < 4 * 3 * 32 * 16; idx += blockDim.x) {
    int j = idx & 15, l = (idx >> 4) & 31, gs = idx >> 9;
    int s = gs % 3, g = gs / 3;
    int v = j >> 1, half = j & 1;
    int kl = (l >= 16 ? 16 : 0) + j;
    int k  = s * 32 + kl;
    int m  = l & 15;                                        // M = h
    float val = (k < DIM + HID) ? Wg[(g * HID + m) * (DIM + HID) + k]
              : (k == DIM + HID ? bg[g * HID + m] : 0.0f);  // bias row at k==80
    sWgA[idx] = (__bf16)val;
  }
  for (int idx = tid; idx < 4 * 4 * 32 * 16; idx += blockDim.x) {
    int j = idx & 15, l = (idx >> 4) & 31, gs = idx >> 9;
    int nt = gs & 3, g = gs >> 2;
    int v = j >> 1, half = j & 1;
    int kl = (l >= 16 ? 16 : 0) + j;
    int f  = nt * 16 + (l & 15);                            // M = FF feature
    float val = (kl < HID) ? W1[(g * FFD + f) * HID + kl]
              : (kl == HID ? b1[g * FFD + f] : 0.0f);       // bias row at k==16
    sW1A[idx] = (__bf16)val;
  }
  for (int idx = tid; idx < 4 * 2 * 32 * 16; idx += blockDim.x) {
    int j = idx & 15, l = (idx >> 4) & 31, gs = idx >> 9;
    int s = gs & 1, g = gs >> 1;
    int v = j >> 1, half = j & 1;
    int kl = (l >= 16 ? 16 : 0) + j;
    int k  = s * 32 + kl;
    int m  = l & 15;                                        // M = h
    sW2A[idx] = (__bf16)W2[(g * HID + m) * FFD + k];
  }
  __syncthreads();

  // b2 / theta per VGPR slot r -> feature h = hiOff + r (D layout of ang^T)
  float b2r[4][8], thr[4][8];
#pragma unroll
  for (int g = 0; g < 4; ++g)
#pragma unroll
    for (int r = 0; r < 8; ++r) {
      b2r[g][r] = b2[g * HID + hiOff + r];
      thr[g][r] = theta[g * HID + hiOff + r];
    }

  v8f hx = {}; v8f cx = {};   // lane b holds h = hiOff + r

  for (int t = 0; t < T_SEQ; ++t) {
    // ---- comb^T B-fragments (contiguous k order used here: lo lanes k 0..15, hi lanes k 16..31) ----
    // s=0,1: x columns -> each lane reads 16 contiguous floats
    FragU xB[2];
    const float* xr = inputs + ((size_t)t * BATCH + rowBase + b) * DIM + (hi ? 16 : 0);
#pragma unroll
    for (int s = 0; s < 2; ++s)
#pragma unroll
      for (int j = 0; j < 4; ++j) {
        float4 w = *(const float4*)(xr + s * 32 + 4 * j);
        xB[s].u[2 * j]     = pkbf(w.x, w.y);
        xB[s].u[2 * j + 1] = pkbf(w.z, w.w);
      }
    // s=2: [hx(16) | 1.0 bias | 0 pad] -- hx half-exchange, all in registers
    FragU hB;
    {
      unsigned int pk[4], rc[4];
#pragma unroll
      for (int j = 0; j < 4; ++j) pk[j] = pkbf(hx[2 * j], hx[2 * j + 1]);
#pragma unroll
      for (int j = 0; j < 4; ++j) rc[j] = __shfl_xor(pk[j], 16, 32);
#pragma unroll
      for (int j = 0; j < 4; ++j) {
        hB.u[j]     = hi ? (j == 0 ? BF16_ONE_LO : 0u) : pk[j]; // hi: bias row k==80 then pad
        hB.u[4 + j] = hi ? 0u : rc[j];
      }
    }

    float iv[8];   // input-gate values, alive only between gate 1 and gate 2
#pragma unroll
    for (int g = 0; g < 4; ++g) {
      // ---- pre^T = Wg . comb^T  (+bg via bias row), 3 WMMAs ----
      v8f pre = {};
      pre = WMMA_BF16(*(const v16bf*)(sWgA + ((g * 3 + 0) * 32 + lane) * 16), xB[0].v, pre);
      pre = WMMA_BF16(*(const v16bf*)(sWgA + ((g * 3 + 1) * 32 + lane) * 16), xB[1].v, pre);
      pre = WMMA_BF16(*(const v16bf*)(sWgA + ((g * 3 + 2) * 32 + lane) * 16), hB.v,    pre);

      // pre^T D -> pre B-fragment (half exchange; hi lanes carry bias row k==16)
      FragU pB;
      {
        unsigned int pk[4], rc[4];
#pragma unroll
        for (int j = 0; j < 4; ++j) pk[j] = pkbf(pre[2 * j], pre[2 * j + 1]);
#pragma unroll
        for (int j = 0; j < 4; ++j) rc[j] = __shfl_xor(pk[j], 16, 32);
#pragma unroll
        for (int j = 0; j < 4; ++j) {
          pB.u[j]     = hi ? (j == 0 ? BF16_ONE_LO : 0u) : pk[j];
          pB.u[4 + j] = hi ? 0u : rc[j];
        }
      }

      // ---- h1^T = relu(W1 . pre^T + b1), 4 f-tiles ----
      unsigned int h1pk[4][4];
#pragma unroll
      for (int nt = 0; nt < 4; ++nt) {
        v8f a1 = {};
        a1 = WMMA_BF16(*(const v16bf*)(sW1A + ((g * 4 + nt) * 32 + lane) * 16), pB.v, a1);
#pragma unroll
        for (int j = 0; j < 4; ++j) {
          float e0 = a1[2 * j],     r0 = e0 > 0.0f ? e0 : 0.0f;
          float e1 = a1[2 * j + 1], r1 = e1 > 0.0f ? e1 : 0.0f;
          h1pk[nt][j] = pkbf(r0, r1);
        }
      }

      // ---- ang^T = W2 . h1^T, K=64 in 2 ksteps; B-frag per kstep via tile-pair exchange --
      v8f a2 = {};
#pragma unroll
      for (int s = 0; s < 2; ++s) {
        FragU bB;
#pragma unroll
        for (int j = 0; j < 4; ++j) {
          // lo lanes want partner's tile(2s) halves; hi lanes want partner's tile(2s+1)
          unsigned int src = hi ? h1pk[2 * s][j] : h1pk[2 * s + 1][j];
          unsigned int rec = __shfl_xor(src, 16, 32);
          bB.u[j]     = hi ? rec : h1pk[2 * s][j];
          bB.u[4 + j] = hi ? h1pk[2 * s + 1][j] : rec;
        }
        a2 = WMMA_BF16(*(const v16bf*)(sW2A + ((g * 2 + s) * 32 + lane) * 16), bB.v, a2);
      }

      // ---- phi = tanh(ang + b2) + theta ; z = cumprod(cos(phi)) along h ----
      // h lives on VGPRs: serial prefix + single cross-half shuffle
      float p[8];
      p[0] = __cosf(fast_tanh(a2[0] + b2r[g][0]) + thr[g][0]);
#pragma unroll
      for (int r = 1; r < 8; ++r)
        p[r] = p[r - 1] * __cosf(fast_tanh(a2[r] + b2r[g][r]) + thr[g][r]);
      float carry = __shfl_xor(p[7], 16, 32);   // hi lanes receive lo half's total
#pragma unroll
      for (int r = 0; r < 8; ++r) {
        float z = hi ? p[r] * carry : p[r];
        // consume gate immediately: f->scale cx, i->keep, g->fold into cx, o->produce hx
        if (g == 0) {
          cx[r] = fast_sigmoid(z) * cx[r];
        } else if (g == 1) {
          iv[r] = fast_sigmoid(z);
        } else if (g == 2) {
          cx[r] = cx[r] + iv[r] * fast_tanh(z);
        } else {
          hx[r] = fast_sigmoid(z) * fast_tanh(cx[r]);
        }
      }
    }

    // ---- contiguous 32B output per lane ----
    float* orow = out + ((size_t)t * BATCH + rowBase + b) * HID + hiOff;
    *(float4*)(orow)     = make_float4(hx[0], hx[1], hx[2], hx[3]);
    *(float4*)(orow + 4) = make_float4(hx[4], hx[5], hx[6], hx[7]);
  }

  // ---- final hx, cx appended after outputs ----
  const size_t base = (size_t)T_SEQ * BATCH * HID;
  {
    float* hrow = out + base + (size_t)(rowBase + b) * HID + hiOff;
    *(float4*)(hrow)     = make_float4(hx[0], hx[1], hx[2], hx[3]);
    *(float4*)(hrow + 4) = make_float4(hx[4], hx[5], hx[6], hx[7]);
    float* crow = out + base + (size_t)BATCH * HID + (size_t)(rowBase + b) * HID + hiOff;
    *(float4*)(crow)     = make_float4(cx[0], cx[1], cx[2], cx[3]);
    *(float4*)(crow + 4) = make_float4(cx[4], cx[5], cx[6], cx[7]);
  }
}

extern "C" void kernel_launch(void* const* d_in, const int* in_sizes, int n_in,
                              void* d_out, int out_size, void* d_ws, size_t ws_size,
                              hipStream_t stream) {
  const float* inputs = (const float*)d_in[0];
  const float* Wg     = (const float*)d_in[1];
  const float* bg     = (const float*)d_in[2];
  const float* W1     = (const float*)d_in[3];
  const float* b1     = (const float*)d_in[4];
  const float* W2     = (const float*)d_in[5];
  const float* b2     = (const float*)d_in[6];
  const float* theta  = (const float*)d_in[7];
  float* out = (float*)d_out;

  dim3 grid(BATCH / ROWS_PER_BLOCK);   // 128 workgroups
  dim3 block(WAVES_PER_BLOCK * 32);    // 2 waves per block
  qlstm_scan_kernel<<<grid, block, 0, stream>>>(inputs, Wg, bg, W1, b1, W2, b2, theta, out);
}
